// RNN_65824668778827
// MI455X (gfx1250) — matrix operand order repaired and checked by reading, not compiled, not run on hardware
//
#include <hip/hip_runtime.h>

// ---------------------------------------------------------------------------
// Vanilla RNN on MI455X (gfx1250, wave32, WMMA).
//   Phase A: xw[b,t,:] = x[b,t,:] @ W + b      (memory-bound, f16 out in d_ws)
//   Phase B: h_t = tanh(xw_t + h_{t-1} @ U)    (latency-bound scan, 1024 steps)
// All GEMMs use v_wmma_f32_16x16x32_f16 (f32 accumulate, K=64 = 2 chained WMMAs).
// Phase B streams xw through a 16-deep LDS ring fed by GLOBAL_LOAD_ASYNC_TO_LDS
// (ASYNCcnt-tracked), overlapping HBM latency with the serial recurrence.
// ---------------------------------------------------------------------------

typedef __attribute__((ext_vector_type(16))) _Float16 v16h;
typedef __attribute__((ext_vector_type(8)))  _Float16 v8h;
typedef __attribute__((ext_vector_type(8)))  float    v8f;

constexpr int Bb = 512;    // batch
constexpr int Tt = 1024;   // timesteps
constexpr int Dd = 64;     // input dim
constexpr int Uu = 64;     // units
constexpr int BT_TILES = Bb / 16;   // 32 independent 16-row batch tiles
constexpr int HSTR = 72;            // padded LDS row stride (halves) -> conflict-free b128
constexpr int DEPTH = 16;           // async xw staging ring depth (steps)

static __device__ __forceinline__ float fast_tanhf(float x) {
#if __has_builtin(__builtin_amdgcn_tanhf)
  return __builtin_amdgcn_tanhf(x);   // gfx1250 V_TANH_F32 (TRANS unit)
#else
  return tanhf(x);
#endif
}

// One 16B-per-lane async global->LDS transfer (ASYNCcnt-tracked, GV mode).
static __device__ __forceinline__ void async_stage_b128(const _Float16* gsrc,
                                                        _Float16* lds_dst) {
  unsigned           laddr = (unsigned)(size_t)lds_dst;              // addr[31:0] = LDS offset
  unsigned long long gaddr = (unsigned long long)(size_t)gsrc;
  asm volatile("global_load_async_to_lds_b128 %0, %1, off"
               :: "v"(laddr), "v"(gaddr) : "memory");
}

#define S_WAIT_ASYNCCNT(n) asm volatile("s_wait_asynccnt " #n ::: "memory")
#define S_WAIT_DSCNT0()    asm volatile("s_wait_dscnt 0x0"     ::: "memory")

// ---------------------------------------------------------------------------
// Phase A: xw = X @ W + bias, written as f16 in raw WMMA C-fragment layout:
//   xw[((bt*1024 + t)*4 + j)*256 + lane*8 + r]  (j = 16-col tile, r = frag reg)
// so phase B reloads its C operand with one contiguous 16B load per lane.
// ---------------------------------------------------------------------------
__global__ __launch_bounds__(256) void xw_proj_kernel(
    const float* __restrict__ X, const float* __restrict__ W,
    const float* __restrict__ bias, _Float16* __restrict__ xw)
{
  const int lane = threadIdx.x & 31;
  const int wid  = blockIdx.x * (blockDim.x >> 5) + (threadIdx.x >> 5);
  const int nwv  = gridDim.x * (blockDim.x >> 5);

  // B-operand fragments of W (table order, matching A):
  // lane l holds column N = j*16 + l%16; element i -> K = 8*(l/16) + i (i<8), + i+8 (i>=8).
  v16h Wf[2][4];
#pragma unroll
  for (int k = 0; k < 2; ++k) {
    const int K = k * 32 + ((lane < 16) ? 0 : 8);
#pragma unroll
    for (int j = 0; j < 4; ++j) {
      v16h f;
#pragma unroll
      for (int i = 0; i < 16; ++i) f[i] = (_Float16)W[(K + (i < 8 ? i : i + 8)) * Uu + j * 16 + (lane & 15)];
      Wf[k][j] = f;
    }
  }
  // C/D layout (ISA table): lane covers column N = (lane&15) + 16*j for all 8 regs.
  float bj[4];
#pragma unroll
  for (int j = 0; j < 4; ++j) bj[j] = bias[(lane & 15) + 16 * j];

  const int row = lane & 15;            // A layout: lane -> row M = lane%16
  const int kb0 = (lane < 16) ? 0 : 8;  // upper lane half holds K+8 / K+24 groups
  const int TILES = BT_TILES * Tt;      // 32768 (bt, t) tiles

  for (int tile = wid; tile < TILES; tile += nwv) {
    const int bt = tile >> 10;
    const int t  = tile & (Tt - 1);
    const float* prow = X + ((size_t)(bt * 16 + row) * Tt + t) * Dd;

    // A fragments (16x32 f16): reg halves i<8 -> K=kb+i, i>=8 -> K=kb+8+i (ISA table).
    v16h Af[2];
#pragma unroll
    for (int f = 0; f < 2; ++f) {
      const int kb = 32 * f + kb0;
      const float4 x0 = *(const float4*)(prow + kb);
      const float4 x1 = *(const float4*)(prow + kb + 4);
      const float4 x2 = *(const float4*)(prow + kb + 16);
      const float4 x3 = *(const float4*)(prow + kb + 20);
      v16h a;
      a[0] = (_Float16)x0.x;  a[1] = (_Float16)x0.y;  a[2]  = (_Float16)x0.z;  a[3]  = (_Float16)x0.w;
      a[4] = (_Float16)x1.x;  a[5] = (_Float16)x1.y;  a[6]  = (_Float16)x1.z;  a[7]  = (_Float16)x1.w;
      a[8] = (_Float16)x2.x;  a[9] = (_Float16)x2.y;  a[10] = (_Float16)x2.z;  a[11] = (_Float16)x2.w;
      a[12] = (_Float16)x3.x; a[13] = (_Float16)x3.y; a[14] = (_Float16)x3.z;  a[15] = (_Float16)x3.w;
      Af[f] = a;
    }
#pragma unroll
    for (int j = 0; j < 4; ++j) {
      v8f acc;
#pragma unroll
      for (int r = 0; r < 8; ++r) acc[r] = bj[j];
      acc = __builtin_amdgcn_wmma_f32_16x16x32_f16(false, Af[0], false, Wf[0][j],
                                                   (short)0, acc, false, false);
      acc = __builtin_amdgcn_wmma_f32_16x16x32_f16(false, Af[1], false, Wf[1][j],
                                                   (short)0, acc, false, false);
      v8h o;
#pragma unroll
      for (int r = 0; r < 8; ++r) o[r] = (_Float16)acc[r];
      *(v8h*)(xw + ((size_t)tile * 4 + j) * 256 + lane * 8) = o;
    }
  }
}

// ---------------------------------------------------------------------------
// Phase B: per-block = one 16-row batch tile; 4 waves, wave j owns columns
// [16j,16j+16). Per step & wave: 2 chained WMMAs + tanh + LDS h round-trip.
// xw_t staged DEPTH steps ahead via async global->LDS; h double-buffered in
// stride-padded LDS with one barrier per step.
// ---------------------------------------------------------------------------
__global__ __launch_bounds__(128) void rnn_scan_kernel(
    const _Float16* __restrict__ xw, const float* __restrict__ U,
    float* __restrict__ out)
{
  const int lane = threadIdx.x & 31;
  const int j    = threadIdx.x >> 5;   // this wave's 16-column slice
  const int bt   = blockIdx.x;         // batch tile 0..31

  __shared__ __align__(16) _Float16 hbuf[2][16 * HSTR];       // h transpose (4.5 KB)
  __shared__ __align__(16) _Float16 xstage[DEPTH][4][256];    // xw ring   (32 KB)

  // U B-fragments for this wave's slice only (16 VGPRs).
  v16h Uf[2];
#pragma unroll
  for (int k = 0; k < 2; ++k) {
    const int K = k * 32 + ((lane < 16) ? 0 : 8);
    v16h f;
#pragma unroll
    for (int i = 0; i < 16; ++i) f[i] = (_Float16)U[(K + (i < 8 ? i : i + 8)) * Uu + j * 16 + (lane & 15)];
    Uf[k] = f;
  }

  const int row   = lane & 15;
  const int kb0   = (lane < 16) ? 0 : 8;
  const int mbase = (lane < 16) ? 0 : 8;   // D layout: reg r -> row mbase+r
  const int n0    = lane & 15;             // D layout: lane -> column n0+16j

  v16h hA[2];                              // h_{t-1} as A fragments; h_0 = 0
#pragma unroll
  for (int f = 0; f < 2; ++f)
#pragma unroll
    for (int i = 0; i < 16; ++i) hA[f][i] = (_Float16)0.0f;

  v8f hlast;
  const _Float16* cwave = xw + ((size_t)bt * Tt * 4 + j) * 256 + lane * 8;
  _Float16*       mystg = &xstage[0][j][lane * 8];   // 256-half stride between slots*4? no: slot stride below

  // Prime the ring: DEPTH in-flight async transfers (ASYNCcnt = DEPTH).
#pragma unroll
  for (int p = 0; p < DEPTH; ++p)
    async_stage_b128(cwave + (size_t)p * 1024, &xstage[p][j][lane * 8]);

  for (int t = 0; t < Tt; ++t) {
    // Async loads complete in order: <= DEPTH-1 outstanding => slot t resident.
    S_WAIT_ASYNCCNT(0xf);
    const v8h cv = *(const v8h*)(&xstage[t & (DEPTH - 1)][j][lane * 8]);
    v8f c;
#pragma unroll
    for (int r = 0; r < 8; ++r) c[r] = (float)cv[r];

    // Refill the slot just consumed with step t+DEPTH (read completed first).
    if (t + DEPTH < Tt) {
      S_WAIT_DSCNT0();
      async_stage_b128(cwave + (size_t)(t + DEPTH) * 1024,
                       &xstage[t & (DEPTH - 1)][j][lane * 8]);
    }

    c = __builtin_amdgcn_wmma_f32_16x16x32_f16(false, hA[0], false, Uf[0],
                                               (short)0, c, false, false);
    c = __builtin_amdgcn_wmma_f32_16x16x32_f16(false, hA[1], false, Uf[1],
                                               (short)0, c, false, false);
#pragma unroll
    for (int r = 0; r < 8; ++r) c[r] = fast_tanhf(c[r]);
    hlast = c;

    // Publish h_t (D layout -> row-major LDS), one buffer per step parity.
    _Float16* hb = &hbuf[t & 1][0];
#pragma unroll
    for (int r = 0; r < 8; ++r)
      hb[(mbase + r) * HSTR + (n0 + 16 * j)] = (_Float16)c[r];
    __syncthreads();   // waits DScnt; one barrier/step is safe with double buffer

    // Reload full 16x64 h_t as A fragments (2x b128 per fragment, conflict-free).
    const _Float16* hr = &hbuf[t & 1][0];
#pragma unroll
    for (int f = 0; f < 2; ++f) {
      const int kb = 32 * f + kb0;
      const v8h lo = *(const v8h*)(hr + row * HSTR + kb);
      const v8h hi = *(const v8h*)(hr + row * HSTR + kb + 16);
      v16h a;
#pragma unroll
      for (int i = 0; i < 8; ++i) { a[i] = lo[i]; a[8 + i] = hi[i]; }
      hA[f] = a;
    }
  }
  (void)mystg;

  // h_last (f32 accumulators, pre-rounding) -> out[512][64]
#pragma unroll
  for (int r = 0; r < 8; ++r)
    out[(size_t)(bt * 16 + mbase + r) * Uu + (n0 + 16 * j)] = hlast[r];
}

// ---------------------------------------------------------------------------
extern "C" void kernel_launch(void* const* d_in, const int* in_sizes, int n_in,
                              void* d_out, int out_size, void* d_ws, size_t ws_size,
                              hipStream_t stream)
{
  (void)in_sizes; (void)n_in; (void)out_size; (void)ws_size;
  const float* X    = (const float*)d_in[0];   // [512,1024,64] f32
  const float* W    = (const float*)d_in[1];   // [64,64] f32
  const float* U    = (const float*)d_in[2];   // [64,64] f32
  const float* bias = (const float*)d_in[3];   // [64] f32
  float* out        = (float*)d_out;           // [512,1,64] f32
  _Float16* xw      = (_Float16*)d_ws;         // 512*1024*64 f16 = 64 MiB scratch

  xw_proj_kernel<<<512, 256, 0, stream>>>(X, W, bias, xw);
  rnn_scan_kernel<<<BT_TILES, 128, 0, stream>>>(xw, U, out);
}
